// RelPropMultiheadSelfAttention_4466765988485
// MI455X (gfx1250) — matrix operand order repaired and checked by reading, not compiled, not run on hardware
//
#include <hip/hip_runtime.h>
#include <stdint.h>

#define IN_DIM  768
#define ATT_DIM 512
#define NHEADS  8
#define HDIM    64
#define SEQ     2048
#define BATCH   4
#define TOKENS  (BATCH * SEQ)      /* 8192 */
#define QKV_DIM (3 * ATT_DIM)      /* 1536 */

typedef __bf16 v16bf __attribute__((ext_vector_type(16)));
typedef float  v8f   __attribute__((ext_vector_type(8)));

union Frag16 { v16bf v; uint32_t u[8]; };

__device__ __forceinline__ uint16_t f2bf(float f) {
  union { float f; uint32_t u; } c; c.f = f;
  uint32_t r = c.u + 0x7FFFu + ((c.u >> 16) & 1u);   // round-to-nearest-even
  return (uint16_t)(r >> 16);
}

// A fragment (16x32, bf16): lane m=lane&15, group g=lane>>4.
// VGPR i holds K pair starting at k = 2i + 8g + (i>=4 ? 8 : 0). Source is
// row-major with K contiguous, so each pair is one aligned dword load.
__device__ __forceinline__ void load_a_frag(Frag16& a, const uint16_t* p, int ld, int lane) {
  const int g = lane >> 4, m = lane & 15;
  const uint16_t* r = p + m * ld;
#pragma unroll
  for (int i = 0; i < 8; ++i) {
    const int k = 2 * i + 8 * g + ((i >= 4) ? 8 : 0);
    a.u[i] = *reinterpret_cast<const uint32_t*>(r + k);
  }
}

// B fragment (32x16, bf16): lane n=lane&15, group g=lane>>4.
// VGPR i holds K pair k = 2i + 8g + (i>=4 ? 8 : 0). Source B[k][n] = src[n*ld + k] (K contiguous).
__device__ __forceinline__ void load_b_frag(Frag16& b, const uint16_t* p, int ld, int lane) {
  const int g = lane >> 4, n = lane & 15;
  const uint16_t* r = p + n * ld;
#pragma unroll
  for (int i = 0; i < 8; ++i) {
    const int k = 2 * i + 8 * g + ((i >= 4) ? 8 : 0);
    b.u[i] = *reinterpret_cast<const uint32_t*>(r + k);
  }
}

__device__ __forceinline__ v8f wmma_bf16(const Frag16& a, const Frag16& b, v8f c) {
  return __builtin_amdgcn_wmma_f32_16x16x32_bf16(false, a.v, false, b.v,
                                                 (short)0, c, false, false);
}

// ---------------------------------------------------------------- convert
__global__ void mha_cvt_bf16(const float* __restrict__ src, uint16_t* __restrict__ dst, int n) {
  int i = blockIdx.x * blockDim.x + threadIdx.x;
  if (i < n) dst[i] = f2bf(src[i]);
}

// ---------------------------------------------------------------- QKV GEMM
// C[m,c] = sum_f X[m,f] * Wqkv[c,f]   (NT GEMM, K=768)
// Epilogue scatters into Q [bh,n,d] (pre-scaled 1/8), K [bh,n,d], V^T [bh,d,n].
__global__ __launch_bounds__(128) void mha_qkv_gemm(
    const uint16_t* __restrict__ Xb,    // [8192,768] bf16
    const uint16_t* __restrict__ Wb,    // [1536,768] bf16
    uint16_t* __restrict__ Qb,          // [32,2048,64]
    uint16_t* __restrict__ Kb,          // [32,2048,64]
    uint16_t* __restrict__ Vtb)         // [32,64,2048]
{
  const int lane = threadIdx.x & 31;
  const int w    = threadIdx.x >> 5;          // 4 waves, 2x2
  const int row0 = blockIdx.x * 64 + (w >> 1) * 32;
  const int col0 = blockIdx.y * 64 + (w & 1) * 32;

  v8f acc[2][2];
#pragma unroll
  for (int i = 0; i < 2; ++i)
#pragma unroll
    for (int j = 0; j < 2; ++j)
      acc[i][j] = (v8f){0.f, 0.f, 0.f, 0.f, 0.f, 0.f, 0.f, 0.f};

  for (int kk = 0; kk < IN_DIM; kk += 32) {
    if (kk + 32 < IN_DIM) {
      __builtin_prefetch(Xb + (size_t)row0 * IN_DIM + kk + 32, 0, 1);
      __builtin_prefetch(Wb + (size_t)col0 * IN_DIM + kk + 32, 0, 1);
    }
    Frag16 a0, a1, b0, b1;
    load_a_frag(a0, Xb + (size_t)row0 * IN_DIM + kk, IN_DIM, lane);
    load_a_frag(a1, Xb + (size_t)(row0 + 16) * IN_DIM + kk, IN_DIM, lane);
    load_b_frag(b0, Wb + (size_t)col0 * IN_DIM + kk, IN_DIM, lane);
    load_b_frag(b1, Wb + (size_t)(col0 + 16) * IN_DIM + kk, IN_DIM, lane);
    acc[0][0] = wmma_bf16(a0, b0, acc[0][0]);
    acc[0][1] = wmma_bf16(a0, b1, acc[0][1]);
    acc[1][0] = wmma_bf16(a1, b0, acc[1][0]);
    acc[1][1] = wmma_bf16(a1, b1, acc[1][1]);
  }

  const int g = lane >> 4, nl = lane & 15;
#pragma unroll
  for (int i = 0; i < 2; ++i)
#pragma unroll
    for (int j = 0; j < 2; ++j)
#pragma unroll
      for (int e = 0; e < 8; ++e) {
        const int m = row0 + 16 * i + e + 8 * g;     // token row
        const int c = col0 + 16 * j + nl;            // qkv column
        const float vv = acc[i][j][e];
        const int b = m >> 11, n = m & (SEQ - 1);
        if (c < ATT_DIM) {
          const int h = c >> 6, dd = c & 63;
          Qb[((size_t)(b * NHEADS + h) * SEQ + n) * HDIM + dd] = f2bf(vv * 0.125f);
        } else if (c < 2 * ATT_DIM) {
          const int c2 = c - ATT_DIM, h = c2 >> 6, dd = c2 & 63;
          Kb[((size_t)(b * NHEADS + h) * SEQ + n) * HDIM + dd] = f2bf(vv);
        } else {
          const int c2 = c - 2 * ATT_DIM, h = c2 >> 6, dd = c2 & 63;
          Vtb[((size_t)(b * NHEADS + h) * HDIM + dd) * SEQ + n] = f2bf(vv);
        }
      }
}

// ---------------------------------------------------------------- attention
// Flash attention: block = (bh, 64 query rows), wave owns 16 rows.
// S and P live in registers; one LDS round-trip per key tile re-layouts P
// (C layout -> A layout). Output written as Y [token, h*64+d] bf16.
__global__ __launch_bounds__(128) void mha_attn(
    const uint16_t* __restrict__ Qb,    // [32,2048,64] pre-scaled
    const uint16_t* __restrict__ Kb,    // [32,2048,64]
    const uint16_t* __restrict__ Vtb,   // [32,64,2048]
    uint16_t* __restrict__ Yb)          // [8192,512]
{
  __shared__ uint16_t pbuf[4][16 * 64];

  const int lane = threadIdx.x & 31;
  const int w    = threadIdx.x >> 5;
  const int bh   = blockIdx.x;                       // 0..31
  const int r0   = blockIdx.y * 64 + w * 16;         // query row in slice
  const int g = lane >> 4, nl = lane & 15;

  // Q fragments for this wave's 16 rows (K = 64 -> two A fragments)
  const uint16_t* Qs = Qb + ((size_t)bh * SEQ + r0) * HDIM;
  Frag16 qa0, qa1;
  load_a_frag(qa0, Qs, HDIM, lane);
  load_a_frag(qa1, Qs + 32, HDIM, lane);

  v8f o[4];
#pragma unroll
  for (int jn = 0; jn < 4; ++jn) o[jn] = (v8f){0.f, 0.f, 0.f, 0.f, 0.f, 0.f, 0.f, 0.f};
  float mrun[8], lrun[8];
#pragma unroll
  for (int e = 0; e < 8; ++e) { mrun[e] = -__builtin_inff(); lrun[e] = 0.f; }

  uint16_t* lp = &pbuf[w][0];

  for (int t = 0; t < SEQ / 64; ++t) {
    const int c0 = t * 64;

    // --- S = Q K^T (16 x 64), f32 accum -------------------------------
    v8f s[4];
#pragma unroll
    for (int jj = 0; jj < 4; ++jj) {
      const uint16_t* Ks = Kb + ((size_t)bh * SEQ + c0 + 16 * jj) * HDIM;
      Frag16 kb0, kb1;
      load_b_frag(kb0, Ks, HDIM, lane);
      load_b_frag(kb1, Ks + 32, HDIM, lane);
      v8f z = (v8f){0.f, 0.f, 0.f, 0.f, 0.f, 0.f, 0.f, 0.f};
      z = wmma_bf16(qa0, kb0, z);
      z = wmma_bf16(qa1, kb1, z);
      s[jj] = z;
    }

    // --- online softmax (row stats across 16-lane column groups) ------
    float sc[8], rsum[8];
#pragma unroll
    for (int e = 0; e < 8; ++e) {
      float v = fmaxf(fmaxf(s[0][e], s[1][e]), fmaxf(s[2][e], s[3][e]));
      v = fmaxf(v, __shfl_xor(v, 1));
      v = fmaxf(v, __shfl_xor(v, 2));
      v = fmaxf(v, __shfl_xor(v, 4));
      v = fmaxf(v, __shfl_xor(v, 8));
      const float mnew = fmaxf(mrun[e], v);
      sc[e] = __expf(mrun[e] - mnew);
      mrun[e] = mnew;
      float rs = 0.f;
#pragma unroll
      for (int jj = 0; jj < 4; ++jj) {
        const float p = __expf(s[jj][e] - mnew);
        s[jj][e] = p;
        rs += p;
      }
      rs += __shfl_xor(rs, 1);
      rs += __shfl_xor(rs, 2);
      rs += __shfl_xor(rs, 4);
      rs += __shfl_xor(rs, 8);
      rsum[e] = rs;
      lrun[e] = lrun[e] * sc[e] + rs;
    }
#pragma unroll
    for (int jn = 0; jn < 4; ++jn)
#pragma unroll
      for (int e = 0; e < 8; ++e) o[jn][e] *= sc[e];

    // --- re-layout P through LDS: C layout -> A layout -----------------
    __syncthreads();   // prior tile's LDS reads are done
#pragma unroll
    for (int jj = 0; jj < 4; ++jj)
#pragma unroll
      for (int e = 0; e < 8; ++e)
        lp[(e + 8 * g) * 64 + 16 * jj + nl] = f2bf(s[jj][e]);
    __syncthreads();

    Frag16 pa0, pa1;
    load_a_frag(pa0, lp, 64, lane);
    load_a_frag(pa1, lp + 32, 64, lane);

    // --- O += P V  (V^T is K-contiguous along keys) --------------------
#pragma unroll
    for (int jn = 0; jn < 4; ++jn) {
      const uint16_t* Vs = Vtb + ((size_t)bh * HDIM + 16 * jn) * SEQ + c0;
      Frag16 vb0, vb1;
      load_b_frag(vb0, Vs, SEQ, lane);
      load_b_frag(vb1, Vs + 32, SEQ, lane);
      o[jn] = wmma_bf16(pa0, vb0, o[jn]);
      o[jn] = wmma_bf16(pa1, vb1, o[jn]);
    }
  }

  // --- normalize and write Y as [token, h*64+d] bf16 --------------------
  const int b = bh >> 3, h = bh & 7;
#pragma unroll
  for (int jn = 0; jn < 4; ++jn)
#pragma unroll
    for (int e = 0; e < 8; ++e) {
      const int qr = r0 + e + 8 * g;
      const size_t token = (size_t)b * SEQ + qr;
      const int col = h * HDIM + 16 * jn + nl;
      Yb[token * ATT_DIM + col] = f2bf(o[jn][e] / lrun[e]);
    }
}

// ---------------------------------------------------------------- out GEMM
// out[m,c] = sum_a Y[m,a] * Wout[c,a] + bias[c]   (NT GEMM, K=512)
__global__ __launch_bounds__(128) void mha_out_gemm(
    const uint16_t* __restrict__ Yb,    // [8192,512] bf16
    const uint16_t* __restrict__ Wob,   // [768,512] bf16
    const float* __restrict__ bias,     // [768]
    float* __restrict__ out)            // [8192,768] f32
{
  const int lane = threadIdx.x & 31;
  const int w    = threadIdx.x >> 5;
  const int row0 = blockIdx.x * 64 + (w >> 1) * 32;
  const int col0 = blockIdx.y * 64 + (w & 1) * 32;

  v8f acc[2][2];
#pragma unroll
  for (int i = 0; i < 2; ++i)
#pragma unroll
    for (int j = 0; j < 2; ++j)
      acc[i][j] = (v8f){0.f, 0.f, 0.f, 0.f, 0.f, 0.f, 0.f, 0.f};

  for (int kk = 0; kk < ATT_DIM; kk += 32) {
    Frag16 a0, a1, b0, b1;
    load_a_frag(a0, Yb + (size_t)row0 * ATT_DIM + kk, ATT_DIM, lane);
    load_a_frag(a1, Yb + (size_t)(row0 + 16) * ATT_DIM + kk, ATT_DIM, lane);
    load_b_frag(b0, Wob + (size_t)col0 * ATT_DIM + kk, ATT_DIM, lane);
    load_b_frag(b1, Wob + (size_t)(col0 + 16) * ATT_DIM + kk, ATT_DIM, lane);
    acc[0][0] = wmma_bf16(a0, b0, acc[0][0]);
    acc[0][1] = wmma_bf16(a0, b1, acc[0][1]);
    acc[1][0] = wmma_bf16(a1, b0, acc[1][0]);
    acc[1][1] = wmma_bf16(a1, b1, acc[1][1]);
  }

  const int g = lane >> 4, nl = lane & 15;
#pragma unroll
  for (int i = 0; i < 2; ++i)
#pragma unroll
    for (int j = 0; j < 2; ++j)
#pragma unroll
      for (int e = 0; e < 8; ++e) {
        const int m = row0 + 16 * i + e + 8 * g;
        const int c = col0 + 16 * j + nl;
        out[(size_t)m * IN_DIM + c] = acc[i][j][e] + bias[c];
      }
}

// ---------------------------------------------------------------- launcher
extern "C" void kernel_launch(void* const* d_in, const int* in_sizes, int n_in,
                              void* d_out, int out_size, void* d_ws, size_t ws_size,
                              hipStream_t stream) {
  const float* X    = (const float*)d_in[0];   // [4,2048,768]
  const float* Wqkv = (const float*)d_in[1];   // [1536,768]
  const float* Wout = (const float*)d_in[2];   // [768,512]
  const float* bout = (const float*)d_in[3];   // [768]
  float* out = (float*)d_out;

  char* base = (char*)d_ws;
  size_t off = 0;
  auto take = [&](size_t nbytes) -> void* {
    void* p = base + off;
    off += (nbytes + 255) & ~(size_t)255;
    return p;
  };

  uint16_t* Xb  = (uint16_t*)take((size_t)TOKENS * IN_DIM * 2);
  uint16_t* Wb  = (uint16_t*)take((size_t)QKV_DIM * IN_DIM * 2);
  uint16_t* Wob = (uint16_t*)take((size_t)IN_DIM * ATT_DIM * 2);
  uint16_t* Qb  = (uint16_t*)take((size_t)BATCH * NHEADS * SEQ * HDIM * 2);
  uint16_t* Kb  = (uint16_t*)take((size_t)BATCH * NHEADS * SEQ * HDIM * 2);
  uint16_t* Vtb = (uint16_t*)take((size_t)BATCH * NHEADS * SEQ * HDIM * 2);
  uint16_t* Yb  = (uint16_t*)take((size_t)TOKENS * ATT_DIM * 2);

  const int nX = TOKENS * IN_DIM;
  const int nWq = QKV_DIM * IN_DIM;
  const int nWo = IN_DIM * ATT_DIM;
  mha_cvt_bf16<<<(nX + 255) / 256, 256, 0, stream>>>(X, Xb, nX);
  mha_cvt_bf16<<<(nWq + 255) / 256, 256, 0, stream>>>(Wqkv, Wb, nWq);
  mha_cvt_bf16<<<(nWo + 255) / 256, 256, 0, stream>>>(Wout, Wob, nWo);

  mha_qkv_gemm<<<dim3(TOKENS / 64, QKV_DIM / 64), 128, 0, stream>>>(Xb, Wb, Qb, Kb, Vtb);
  mha_attn<<<dim3(BATCH * NHEADS, SEQ / 64), 128, 0, stream>>>(Qb, Kb, Vtb, Yb);
  mha_out_gemm<<<dim3(TOKENS / 64, IN_DIM / 64), 128, 0, stream>>>(Yb, Wob, bout, out);
}
